// ExtraRelation_60945585930504
// MI455X (gfx1250) — matrix operand order repaired and checked by reading, not compiled, not run on hardware
//
#include <hip/hip_runtime.h>
#include <hip/hip_bf16.h>
#include <math.h>

typedef __bf16 bf16;
typedef __attribute__((ext_vector_type(16))) __bf16 v16bf;
typedef __attribute__((ext_vector_type(8)))  __bf16 bf16x8;
typedef __attribute__((ext_vector_type(8)))  float  v8f;

namespace cfg {
constexpr int B    = 4;
constexpr int S    = 512;
constexpr int H    = 768;
constexpr int NH   = 12;
constexpr int E    = 24;
constexpr int P    = 552;          // E*(E-1) ordered pairs
constexpr int PP   = 560;          // P padded to multiple of 16
constexpr int TE   = 20;
constexpr int TD   = 20;
constexpr int R    = 97;
constexpr int RP   = 112;          // R padded to multiple of 16
constexpr int DIN  = 1576;         // 2H + TE + TD
constexpr int DINP = 1600;         // padded to multiple of 32
constexpr int L    = B * P;        // 2208 (multiple of 16)
constexpr int KC   = 12 * 64 * 64; // 49152 contraction dim of logits GEMM
}

// ---------------------------------------------------------------------------
// WMMA helpers (CDNA5 wave32, v_wmma_f32_16x16x32_bf16)
// ---------------------------------------------------------------------------
__device__ __forceinline__ v8f wmma_bf16(v16bf a, v16bf b, v8f c) {
#if defined(__HIP_DEVICE_COMPILE__)
  // 8 args: (neg_a, A, neg_b, B, c_mod, C, reuse_a, reuse_b)
  return __builtin_amdgcn_wmma_f32_16x16x32_bf16(false, a, false, b, (short)0, c,
                                                 false, false);
#else
  return c;
#endif
}

// A-matrix 16x32 bf16 layout: lane holds row M=lane%16.
//   a[0..7]  <-> K = 8*half + 0..7      (half = lane/16)
//   a[8..15] <-> K = 16 + 8*half + 0..7
// p = &A[row][k0]  (row-major A)
__device__ __forceinline__ v16bf load_a_frag(const bf16* __restrict__ p, int half) {
  union { bf16x8 h[2]; v16bf v; } u;
  u.h[0] = *(const bf16x8*)(p + half * 8);
  u.h[1] = *(const bf16x8*)(p + 16 + half * 8);
  return u.v;
}

// B-matrix 32x16 bf16 layout: lane holds column N=lane%16, b[0..7] <-> K = 8*half + 0..7,
// b[8..15] <-> K = 16 + 8*half + 0..7.  p = &Bt[col][k0]  (Bt = B transposed, N x K row-major)
__device__ __forceinline__ v16bf load_b_frag(const bf16* __restrict__ p, int half) {
  union { bf16x8 h[2]; v16bf v; } u;
  u.h[0] = *(const bf16x8*)(p + half * 8);
  u.h[1] = *(const bf16x8*)(p + 16 + half * 8);
  return u.v;
}

// ---------------------------------------------------------------------------
// Stage 1: per-(b,e) head-sum of the two gathered attention rows + entity feats
// ent_att[b,e,s] = 0.5 * sum_nh (attn[b,nh,hd,s] + attn[b,nh,tl,s])
// ent_feat[b,e,:] = 0.5*(h[hd]+h[tl])   (stored bf16, only feeds bf16 GEMM)
// ---------------------------------------------------------------------------
__global__ void k_ent(const float* __restrict__ hidden,
                      const float* __restrict__ attn,
                      const int* __restrict__ head, const int* __restrict__ tail,
                      float* __restrict__ ent_att, bf16* __restrict__ ent_feat) {
  const int be = blockIdx.x;              // b*E + e
  const int b  = be / cfg::E;
  const int hd = head[be], tl = tail[be];
  const size_t attB = (size_t)b * cfg::NH * cfg::S * cfg::S;
  for (int s = threadIdx.x; s < cfg::S; s += blockDim.x) {
    float acc = 0.f;
    for (int nh = 0; nh < cfg::NH; ++nh) {
      const float* a = attn + attB + (size_t)nh * cfg::S * cfg::S;
      acc += a[(size_t)hd * cfg::S + s] + a[(size_t)tl * cfg::S + s];
    }
    ent_att[(size_t)be * cfg::S + s] = 0.5f * acc;
  }
  const float* hb = hidden + (size_t)b * cfg::S * cfg::H;
  for (int j = threadIdx.x; j < cfg::H; j += blockDim.x) {
    ent_feat[(size_t)be * cfg::H + j] =
        (bf16)(0.5f * (hb[(size_t)hd * cfg::H + j] + hb[(size_t)tl * cfg::H + j]));
  }
}

// ---------------------------------------------------------------------------
// Stage 2: normalized pair attention rows -> A1 (B, PP, S) bf16 (GEMM1 A)
// ---------------------------------------------------------------------------
__global__ void k_htatt(const float* __restrict__ ent_att,
                        const float* __restrict__ mask,
                        bf16* __restrict__ A1) {
  const int bp = blockIdx.x;              // b*PP + p
  const int b  = bp / cfg::PP;
  const int p  = bp % cfg::PP;
  bf16* row = A1 + (size_t)bp * cfg::S;
  if (p >= cfg::P) {                      // zero padding rows
    for (int s = threadIdx.x; s < cfg::S; s += blockDim.x) row[s] = (bf16)0.f;
    return;
  }
  const int i0 = p / (cfg::E - 1);
  const int j  = p % (cfg::E - 1);
  const int i1 = j + (j >= i0 ? 1 : 0);   // itertools.permutations order
  const float* e0 = ent_att + ((size_t)b * cfg::E + i0) * cfg::S;
  const float* e1 = ent_att + ((size_t)b * cfg::E + i1) * cfg::S;
  const float* m  = mask + (size_t)b * cfg::S;

  __shared__ float red[256];
  float part = 0.f;
  for (int s = threadIdx.x; s < cfg::S; s += blockDim.x)
    part += e0[s] * e1[s] * m[s];
  red[threadIdx.x] = part;
  __syncthreads();
  for (int off = blockDim.x >> 1; off > 0; off >>= 1) {
    if ((int)threadIdx.x < off) red[threadIdx.x] += red[threadIdx.x + off];
    __syncthreads();
  }
  const float inv = 1.f / (red[0] + 1e-20f);
  for (int s = threadIdx.x; s < cfg::S; s += blockDim.x)
    row[s] = (bf16)(e0[s] * e1[s] * m[s] * inv);
}

// ---------------------------------------------------------------------------
// Conversions / transposes into bf16 workspace
// ---------------------------------------------------------------------------
__global__ void k_hT(const float* __restrict__ hidden, bf16* __restrict__ hT,
                     int total) {   // hT[b][n][s] = hidden[b][s][n]
  int idx = blockIdx.x * blockDim.x + threadIdx.x;
  if (idx >= total) return;
  int s = idx % cfg::S;
  int n = (idx / cfg::S) % cfg::H;
  int b = idx / (cfg::S * cfg::H);
  hT[idx] = (bf16)hidden[((size_t)b * cfg::S + s) * cfg::H + n];
}

__global__ void k_convW(const float* __restrict__ W, bf16* __restrict__ WT,
                        int total) {  // WT[n][k] = W[k][n], K zero-padded
  int idx = blockIdx.x * blockDim.x + threadIdx.x;
  if (idx >= total) return;
  int k = idx % cfg::DINP;
  int n = idx / cfg::DINP;
  WT[idx] = (k < cfg::DIN) ? (bf16)W[(size_t)k * cfg::H + n] : (bf16)0.f;
}

__global__ void k_convCls(const float* __restrict__ clsW, bf16* __restrict__ out,
                          int total) { // out[r][k] = clsW[k-th (n,b,c)][r], R padded
  int idx = blockIdx.x * blockDim.x + threadIdx.x;
  if (idx >= total) return;
  int k = idx % cfg::KC;
  int r = idx / cfg::KC;
  out[idx] = (r < cfg::R) ? (bf16)clsW[(size_t)k * cfg::R + r] : (bf16)0.f;
}

// ---------------------------------------------------------------------------
// GEMM1: ht_info[b] = A1[b] (PPxS) @ h[b] (SxH); Bt = hT[b] (HxS)
// one wave computes a 16x64 tile (A frag reused across 4 WMMAs)
// ---------------------------------------------------------------------------
__global__ __launch_bounds__(32)
void k_gemm_htinfo(const bf16* __restrict__ A1, const bf16* __restrict__ hT,
                   bf16* __restrict__ htinfo) {
  const int lane = threadIdx.x, half = lane >> 4, l16 = lane & 15;
  const int n0 = blockIdx.x * 64;
  const int m0 = blockIdx.y * 16;
  const int b  = blockIdx.z;
  const bf16* Arow = A1 + ((size_t)b * cfg::PP + m0 + l16) * cfg::S;
  const bf16* Bt   = hT + (size_t)b * cfg::H * cfg::S;
  const bf16* B0 = Bt + (size_t)(n0 +  0 + l16) * cfg::S;
  const bf16* B1 = Bt + (size_t)(n0 + 16 + l16) * cfg::S;
  const bf16* B2 = Bt + (size_t)(n0 + 32 + l16) * cfg::S;
  const bf16* B3 = Bt + (size_t)(n0 + 48 + l16) * cfg::S;
  v8f acc0 = {}, acc1 = {}, acc2 = {}, acc3 = {};
  for (int k0 = 0; k0 < cfg::S; k0 += 32) {
    __builtin_prefetch(Arow + k0 + 128, 0, 1);
    v16bf a  = load_a_frag(Arow + k0, half);
    acc0 = wmma_bf16(a, load_b_frag(B0 + k0, half), acc0);
    acc1 = wmma_bf16(a, load_b_frag(B1 + k0, half), acc1);
    acc2 = wmma_bf16(a, load_b_frag(B2 + k0, half), acc2);
    acc3 = wmma_bf16(a, load_b_frag(B3 + k0, half), acc3);
  }
  bf16* out = htinfo + (size_t)b * cfg::PP * cfg::H;
  v8f accs[4] = {acc0, acc1, acc2, acc3};
  for (int t = 0; t < 4; ++t) {
    const int col = n0 + 16 * t + l16;
    for (int rr = 0; rr < 8; ++rr) {
      const int row = m0 + rr + 8 * half;     // C/D layout: VGPR r -> M=r(+8)
      out[(size_t)row * cfg::H + col] = (bf16)accs[t][rr];
    }
  }
}

// ---------------------------------------------------------------------------
// Pack all_h / all_t (L x DINP) bf16 = [feat | ht_info | type | dist | 0-pad]
// ---------------------------------------------------------------------------
__global__ void k_pack(const bf16* __restrict__ ent_feat,
                       const bf16* __restrict__ htinfo,
                       const int* __restrict__ head, const int* __restrict__ tail,
                       const int* __restrict__ etype,
                       const float* __restrict__ type_table,
                       const float* __restrict__ dis_table,
                       bf16* __restrict__ all_h, bf16* __restrict__ all_t) {
  const int l = blockIdx.x;
  const int b = l / cfg::P;
  const int p = l % cfg::P;
  const int i0 = p / (cfg::E - 1);
  const int j  = p % (cfg::E - 1);
  const int i1 = j + (j >= i0 ? 1 : 0);
  int dt = tail[b * cfg::E + i0] - head[b * cfg::E + i1];
  if (dt < 0) dt = -dt;
  int dbkt = 0;
  {
    const int th[9] = {2, 4, 8, 16, 32, 64, 128, 256, 512};
    for (int i = 0; i < 9; ++i) if (dt >= th[i]) dbkt = i + 1;
  }
  const int et0 = etype[b * cfg::E + i0];
  const int et1 = etype[b * cfg::E + i1];
  const bf16* f0 = ent_feat + ((size_t)b * cfg::E + i0) * cfg::H;
  const bf16* f1 = ent_feat + ((size_t)b * cfg::E + i1) * cfg::H;
  const bf16* hi = htinfo + ((size_t)b * cfg::PP + p) * cfg::H;
  bf16* oh = all_h + (size_t)l * cfg::DINP;
  bf16* ot = all_t + (size_t)l * cfg::DINP;
  for (int c = threadIdx.x; c < cfg::DINP; c += blockDim.x) {
    bf16 vh, vt;
    if (c < cfg::H)               { vh = f0[c]; vt = f1[c]; }
    else if (c < 2 * cfg::H)      { bf16 v = hi[c - cfg::H]; vh = v; vt = v; }
    else if (c < 2 * cfg::H + cfg::TE) {
      const int q = c - 2 * cfg::H;
      vh = (bf16)type_table[et0 * cfg::TE + q];
      vt = (bf16)type_table[et1 * cfg::TE + q];
    } else if (c < cfg::DIN) {
      const int q = c - (2 * cfg::H + cfg::TE);
      bf16 v = (bf16)dis_table[dbkt * cfg::TD + q];
      vh = v; vt = v;
    } else { vh = (bf16)0.f; vt = (bf16)0.f; }
    oh[c] = vh; ot[c] = vt;
  }
}

// ---------------------------------------------------------------------------
// GEMM2: x = tanh(all (LxDINP) @ W (DINPxH) + bias); Bt = WT (HxDINP)
// ---------------------------------------------------------------------------
__global__ __launch_bounds__(32)
void k_gemm_tanh(const bf16* __restrict__ A, const bf16* __restrict__ WT,
                 const float* __restrict__ bias, bf16* __restrict__ out) {
  const int lane = threadIdx.x, half = lane >> 4, l16 = lane & 15;
  const int n0 = blockIdx.x * 64;
  const int m0 = blockIdx.y * 16;
  const bf16* Arow = A + (size_t)(m0 + l16) * cfg::DINP;
  const bf16* B0 = WT + (size_t)(n0 +  0 + l16) * cfg::DINP;
  const bf16* B1 = WT + (size_t)(n0 + 16 + l16) * cfg::DINP;
  const bf16* B2 = WT + (size_t)(n0 + 32 + l16) * cfg::DINP;
  const bf16* B3 = WT + (size_t)(n0 + 48 + l16) * cfg::DINP;
  v8f acc0 = {}, acc1 = {}, acc2 = {}, acc3 = {};
  for (int k0 = 0; k0 < cfg::DINP; k0 += 32) {
    __builtin_prefetch(Arow + k0 + 128, 0, 1);
    v16bf a = load_a_frag(Arow + k0, half);
    acc0 = wmma_bf16(a, load_b_frag(B0 + k0, half), acc0);
    acc1 = wmma_bf16(a, load_b_frag(B1 + k0, half), acc1);
    acc2 = wmma_bf16(a, load_b_frag(B2 + k0, half), acc2);
    acc3 = wmma_bf16(a, load_b_frag(B3 + k0, half), acc3);
  }
  v8f accs[4] = {acc0, acc1, acc2, acc3};
  for (int t = 0; t < 4; ++t) {
    const int col = n0 + 16 * t + l16;
    const float bb = bias[col];
    for (int rr = 0; rr < 8; ++rr) {
      const int row = m0 + rr + 8 * half;
      out[(size_t)row * cfg::H + col] = (bf16)tanhf(accs[t][rr] + bb);
    }
  }
}

// ---------------------------------------------------------------------------
// Logits: logits[l,r] = sum_k A[l,k]*clsWr[r,k] + clsb[r],
//         A[l, n*4096+b*64+c] = x1[l,n*64+b] * x2[l,n*64+c]  (built on the fly)
// K chunk of 32 never crosses a (n,b) boundary.
// ---------------------------------------------------------------------------
__global__ __launch_bounds__(32)
void k_logits(const bf16* __restrict__ x1, const bf16* __restrict__ x2,
              const bf16* __restrict__ clsWr, const float* __restrict__ clsb,
              float* __restrict__ out) {
  const int lane = threadIdx.x, half = lane >> 4, l16 = lane & 15;
  const int n0 = blockIdx.x * 16;
  const int m0 = blockIdx.y * 16;
  const int m  = m0 + l16;
  const bf16* x1r = x1 + (size_t)m * cfg::H;
  const bf16* x2r = x2 + (size_t)m * cfg::H;
  const bf16* Bt  = clsWr + (size_t)(n0 + l16) * cfg::KC;
  v8f acc = {};
  for (int k0 = 0; k0 < cfg::KC; k0 += 32) {
    const int n  = k0 >> 12;         // / 4096
    const int bb = (k0 & 4095) >> 6; // / 64
    const int c0 = k0 & 63;          // 0 or 32
    const float s1 = (float)x1r[n * 64 + bb];
    const bf16* xp = x2r + n * 64 + c0;
    bf16x8 lo = *(const bf16x8*)(xp + half * 8);
    bf16x8 hi = *(const bf16x8*)(xp + 16 + half * 8);
    union { bf16x8 h[2]; v16bf v; } u;
    for (int i = 0; i < 8; ++i) u.h[0][i] = (bf16)(s1 * (float)lo[i]);
    for (int i = 0; i < 8; ++i) u.h[1][i] = (bf16)(s1 * (float)hi[i]);
    __builtin_prefetch(Bt + k0 + 512, 0, 1);
    acc = wmma_bf16(u.v, load_b_frag(Bt + k0, half), acc);
  }
  const int r = n0 + l16;
  if (r < cfg::R) {
    const float cb = clsb[r];
    for (int rr = 0; rr < 8; ++rr) {
      const int row = m0 + rr + 8 * half;
      out[(size_t)row * cfg::R + r] = acc[rr] + cb;
    }
  }
}

// ---------------------------------------------------------------------------
extern "C" void kernel_launch(void* const* d_in, const int* in_sizes, int n_in,
                              void* d_out, int out_size, void* d_ws, size_t ws_size,
                              hipStream_t stream) {
  using namespace cfg;
  const float* hidden = (const float*)d_in[0];
  const float* attn   = (const float*)d_in[1];
  const float* mask   = (const float*)d_in[2];
  const int*   head   = (const int*)d_in[3];
  const int*   tail   = (const int*)d_in[4];
  const int*   etype  = (const int*)d_in[5];
  const float* ttab   = (const float*)d_in[6];
  const float* dtabl  = (const float*)d_in[7];
  const float* Wh     = (const float*)d_in[8];
  const float* bh     = (const float*)d_in[9];
  const float* Wt     = (const float*)d_in[10];
  const float* bt     = (const float*)d_in[11];
  const float* clsW   = (const float*)d_in[12];
  const float* clsb   = (const float*)d_in[13];
  float* logits = (float*)d_out;

  char* wsp = (char*)d_ws;
  auto alloc = [&](size_t bytes) -> char* {
    char* p = wsp;
    wsp += (bytes + 255) & ~(size_t)255;
    return p;
  };
  float* ent_att  = (float*)alloc(sizeof(float) * B * E * S);
  bf16*  ent_feat = (bf16*) alloc(sizeof(bf16) * B * E * H);
  bf16*  A1       = (bf16*) alloc(sizeof(bf16) * (size_t)B * PP * S);
  bf16*  hT       = (bf16*) alloc(sizeof(bf16) * (size_t)B * H * S);
  bf16*  htinfo   = (bf16*) alloc(sizeof(bf16) * (size_t)B * PP * H);
  bf16*  all_h    = (bf16*) alloc(sizeof(bf16) * (size_t)L * DINP);
  bf16*  all_t    = (bf16*) alloc(sizeof(bf16) * (size_t)L * DINP);
  bf16*  WhT      = (bf16*) alloc(sizeof(bf16) * (size_t)H * DINP);
  bf16*  WtT      = (bf16*) alloc(sizeof(bf16) * (size_t)H * DINP);
  bf16*  h1       = (bf16*) alloc(sizeof(bf16) * (size_t)L * H);
  bf16*  t1       = (bf16*) alloc(sizeof(bf16) * (size_t)L * H);
  bf16*  clsWr    = (bf16*) alloc(sizeof(bf16) * (size_t)RP * KC);
  (void)ws_size; (void)in_sizes; (void)n_in; (void)out_size;

  k_ent<<<B * E, 256, 0, stream>>>(hidden, attn, head, tail, ent_att, ent_feat);
  k_htatt<<<B * PP, 256, 0, stream>>>(ent_att, mask, A1);
  { int tot = B * H * S;  k_hT<<<(tot + 255) / 256, 256, 0, stream>>>(hidden, hT, tot); }
  { int tot = H * DINP;   k_convW<<<(tot + 255) / 256, 256, 0, stream>>>(Wh, WhT, tot);
                          k_convW<<<(tot + 255) / 256, 256, 0, stream>>>(Wt, WtT, tot); }
  { int tot = RP * KC;    k_convCls<<<(tot + 255) / 256, 256, 0, stream>>>(clsW, clsWr, tot); }

  k_gemm_htinfo<<<dim3(H / 64, PP / 16, B), 32, 0, stream>>>(A1, hT, htinfo);
  k_pack<<<L, 256, 0, stream>>>(ent_feat, htinfo, head, tail, etype, ttab, dtabl,
                                all_h, all_t);
  k_gemm_tanh<<<dim3(H / 64, L / 16), 32, 0, stream>>>(all_h, WhT, bh, h1);
  k_gemm_tanh<<<dim3(H / 64, L / 16), 32, 0, stream>>>(all_t, WtT, bt, t1);
  k_logits<<<dim3(RP / 16, L / 16), 32, 0, stream>>>(h1, t1, clsWr, clsb, logits);
}
